// GPT_25692494365366
// MI455X (gfx1250) — matrix operand order repaired and checked by reading, not compiled, not run on hardware
//
#include <hip/hip_runtime.h>
#include <hip/hip_bf16.h>
#include <math.h>
#include <stdint.h>

// ---------------------------------------------------------------------------
// Problem constants (match reference)
// ---------------------------------------------------------------------------
constexpr int NB  = 2;        // batch
constexpr int NT  = 1024;     // seq len
constexpr int ND  = 1024;     // model dim
constexpr int NH  = 16;       // heads
constexpr int NHD = 64;       // head dim
constexpr int NL  = 4;        // layers
constexpr int NV  = 32000;    // vocab

typedef __attribute__((ext_vector_type(16))) __bf16 v16bf;
typedef __attribute__((ext_vector_type(8)))  __bf16 v8bf;
typedef __attribute__((ext_vector_type(8)))  float  v8f;

typedef unsigned int u32x4 __attribute__((ext_vector_type(4)));
typedef int          i32x4 __attribute__((ext_vector_type(4)));
typedef int          i32x8 __attribute__((ext_vector_type(8)));

#if defined(__gfx1250__) && __has_builtin(__builtin_amdgcn_tensor_load_to_lds)
#define USE_TDM 1
#else
#define USE_TDM 0
#endif

__device__ inline v8f zero_v8f() {
  v8f z;
#pragma unroll
  for (int i = 0; i < 8; ++i) z[i] = 0.0f;
  return z;
}

// ---------------------------------------------------------------------------
// TDM: 2D bf16 tile (tile_d0 elems/row x tile_d1 rows) -> LDS, optional row pad.
// D# bitfields per CDNA5 ISA ch.8. lds_addr: low 32 bits of generic shared ptr
// (flat shared addressing keeps the LDS offset in addr[31:0]).
// pad_interval code: pads after 2^(code+1) DWORDs; pad_amount code: (code+1) DWORDs.
// ---------------------------------------------------------------------------
__device__ inline void tdm_load_2d_bf16(
    unsigned lds_addr, const void* gptr,
    unsigned tensor_d0, unsigned tensor_d1, unsigned long long stride_elems,
    unsigned tile_d0, unsigned tile_d1,
    unsigned pad_en, unsigned pad_interval, unsigned pad_amount)
{
#if USE_TDM
  unsigned long long ga = (unsigned long long)(uintptr_t)gptr;
  u32x4 g0;
  g0[0] = 1u;                                              // count=1 (user mode)
  g0[1] = lds_addr;                                        // LDS byte address
  g0[2] = (unsigned)(ga & 0xffffffffu);                    // global_addr[31:0]
  g0[3] = (unsigned)((ga >> 32) & 0x01ffffffu) | (2u << 30); // addr[56:32] | type=2
  i32x8 g1;
  g1[0] = (int)((1u << 16)                                 // data_size=2B
              | (pad_en << 20) | (pad_interval << 22) | (pad_amount << 25));
  g1[1] = (int)((tensor_d0 & 0xffffu) << 16);              // tensor_dim0[15:0]
  g1[2] = (int)(((tensor_d0 >> 16) & 0xffffu)              // tensor_dim0[31:16]
              | ((tensor_d1 & 0xffffu) << 16));            // tensor_dim1[15:0]
  g1[3] = (int)(((tensor_d1 >> 16) & 0xffffu)              // tensor_dim1[31:16]
              | ((tile_d0 & 0xffffu) << 16));              // tile_dim0
  g1[4] = (int)(tile_d1 & 0xffffu);                        // tile_dim1 (tile_dim2=0)
  g1[5] = (int)(unsigned)(stride_elems & 0xffffffffull);   // dim0_stride[31:0]
  g1[6] = (int)(unsigned)((stride_elems >> 32) & 0xffffull); // dim0_stride[47:32]
  g1[7] = 0;                                               // dim1_stride unused (2D)
  i32x4 gz = {0, 0, 0, 0};
#if defined(__clang_major__) && (__clang_major__ >= 23)
  i32x8 gz8 = {0, 0, 0, 0, 0, 0, 0, 0};
  __builtin_amdgcn_tensor_load_to_lds(g0, g1, gz, gz, gz8, 0);
#else
  __builtin_amdgcn_tensor_load_to_lds(g0, g1, gz, gz, 0);
#endif
#else
  (void)lds_addr; (void)gptr; (void)tensor_d0; (void)tensor_d1;
  (void)stride_elems; (void)tile_d0; (void)tile_d1;
  (void)pad_en; (void)pad_interval; (void)pad_amount;
#endif
}

// ---------------------------------------------------------------------------
// Weight prep: transpose + convert f32 [K][N] -> bf16 [N][K]
// ---------------------------------------------------------------------------
__global__ __launch_bounds__(256) void transpose_cvt_kernel(
    const float* __restrict__ src, __bf16* __restrict__ dst, int K, int N)
{
  __shared__ float tile[32][33];
  const int k0 = blockIdx.y * 32;
  const int n0 = blockIdx.x * 32;
  const int tid = threadIdx.x;
  const int tr = tid >> 5;       // 0..7
  const int tc = tid & 31;
#pragma unroll
  for (int i = 0; i < 4; ++i) {
    int k = tr + i * 8;
    tile[k][tc] = src[(size_t)(k0 + k) * N + (n0 + tc)];
  }
  __syncthreads();
#pragma unroll
  for (int i = 0; i < 4; ++i) {
    int n = tr + i * 8;
    dst[(size_t)(n0 + n) * K + (k0 + tc)] = (__bf16)tile[tc][n];
  }
}

// ---------------------------------------------------------------------------
// Embedding: x[b,t,:] = wte[ctx[b,t],:] + wpe[t,:]  (f32 residual stream)
// ---------------------------------------------------------------------------
__global__ __launch_bounds__(256) void embed_kernel(
    const int* __restrict__ ctx, const float* __restrict__ wte,
    const float* __restrict__ wpe, float* __restrict__ x)
{
  size_t i = (size_t)blockIdx.x * 256 + threadIdx.x;
  int d = (int)(i % ND);
  size_t bt = i / ND;
  int t = (int)(bt % NT);
  int tok = ctx[bt];
  x[i] = wte[(size_t)tok * ND + d] + wpe[(size_t)t * ND + d];
}

__global__ __launch_bounds__(256) void add_kernel(
    float* __restrict__ x, const float* __restrict__ o)
{
  size_t i = (size_t)blockIdx.x * 256 + threadIdx.x;
  x[i] += o[i];
}

// ---------------------------------------------------------------------------
// LayerNorm (D=1024), one block per row, bf16 output feeding the WMMA GEMMs
// ---------------------------------------------------------------------------
__global__ __launch_bounds__(256) void layernorm_kernel(
    const float* __restrict__ x, const float* __restrict__ g,
    const float* __restrict__ b, __bf16* __restrict__ y)
{
  const int row = blockIdx.x;
  const int tid = threadIdx.x;
  const float* xr = x + (size_t)row * ND;
  float s = 0.0f, ss = 0.0f;
  for (int i = tid; i < ND; i += 256) {
    float v = xr[i];
    s += v; ss += v * v;
  }
#pragma unroll
  for (int off = 16; off > 0; off >>= 1) {
    s  += __shfl_xor(s,  off, 32);
    ss += __shfl_xor(ss, off, 32);
  }
  __shared__ float s1[8], s2[8];
  if ((tid & 31) == 0) { s1[tid >> 5] = s; s2[tid >> 5] = ss; }
  __syncthreads();
  float ts = 0.0f, tss = 0.0f;
#pragma unroll
  for (int i = 0; i < 8; ++i) { ts += s1[i]; tss += s2[i]; }
  const float mean = ts * (1.0f / ND);
  const float var  = tss * (1.0f / ND) - mean * mean;
  const float rstd = rsqrtf(var + 1e-5f);
  __bf16* yr = y + (size_t)row * ND;
  for (int i = tid; i < ND; i += 256)
    yr[i] = (__bf16)((xr[i] - mean) * rstd * g[i] + b[i]);
}

// ---------------------------------------------------------------------------
// bf16 WMMA GEMM: C[M,N] = A[M,K] * B[K,N], B given pre-transposed as Bt[N][K].
// Block 256 thr / 8 waves, tile 128x64, K-step 32, double-buffered TDM staging.
// Rows padded to 40 elems (80B) -> conflict-free 16-lane LDS fragment reads.
// ---------------------------------------------------------------------------
template <typename OutT, bool BIAS, bool RELU, bool RES>
__global__ __launch_bounds__(256) void gemm_bf16_wmma(
    const __bf16* __restrict__ A,   // [M][K] bf16
    const __bf16* __restrict__ Bt,  // [N][K] bf16 (pre-transposed weight)
    const float* __restrict__ bias, const float* __restrict__ res,
    OutT* __restrict__ C, int M, int N, int K)
{
  __shared__ __attribute__((aligned(16))) __bf16 Asb[2][128][40];
  __shared__ __attribute__((aligned(16))) __bf16 Btb[2][64][40];

  const int tid  = threadIdx.x;
  const int lane = tid & 31;
  const int wv   = tid >> 5;
  const int lm   = lane & 15;
  const int hf   = lane >> 4;
  const int wRow = wv & 3;
  const int wCol = wv >> 2;
  const int m0 = blockIdx.y * 128;
  const int n0 = blockIdx.x * 64;

  v8f acc[2][2];
#pragma unroll
  for (int i = 0; i < 2; ++i)
#pragma unroll
    for (int j = 0; j < 2; ++j) acc[i][j] = zero_v8f();

  auto issue = [&](int buf, int k0) {
#if USE_TDM
    if (wv == 0) {
      // pad: after 16 DWORDs (32 bf16 row) insert 4 DWORDs (8 bf16) -> 40-elem rows
      tdm_load_2d_bf16((unsigned)(uintptr_t)&Asb[buf][0][0],
                       A + (size_t)m0 * K + k0,
                       (unsigned)K, (unsigned)M, (unsigned long long)K,
                       32u, 128u, 1u, 3u, 3u);
      tdm_load_2d_bf16((unsigned)(uintptr_t)&Btb[buf][0][0],
                       Bt + (size_t)n0 * K + k0,
                       (unsigned)K, (unsigned)N, (unsigned long long)K,
                       32u, 64u, 1u, 3u, 3u);
    }
#else
    for (int idx = tid; idx < 128 * 4; idx += 256) {
      int r = idx >> 2, c = (idx & 3) * 8;
      *(v8bf*)&Asb[buf][r][c] = *(const v8bf*)(A + (size_t)(m0 + r) * K + k0 + c);
    }
    for (int idx = tid; idx < 64 * 4; idx += 256) {
      int r = idx >> 2, c = (idx & 3) * 8;
      *(v8bf*)&Btb[buf][r][c] = *(const v8bf*)(Bt + (size_t)(n0 + r) * K + k0 + c);
    }
#endif
  };

  const int nsteps = K >> 5;
  issue(0, 0);
  for (int s = 0; s < nsteps; ++s) {
    const int cur = s & 1;
    const bool more = (s + 1) < nsteps;
    if (more) issue(cur ^ 1, (s + 1) << 5);
#if USE_TDM
    if (wv == 0) {
      if (more) __builtin_amdgcn_s_wait_tensorcnt(2);  // current buffer done
      else      __builtin_amdgcn_s_wait_tensorcnt(0);
    }
#endif
    __syncthreads();

    v16bf afr[2], bfr[2];
#pragma unroll
    for (int ti = 0; ti < 2; ++ti) {
      const __bf16* ar = Asb[cur][wRow * 32 + ti * 16 + lm];
      v8bf lo = *(const v8bf*)(ar + hf * 8);        // K = hf*8 .. +7
      v8bf hi = *(const v8bf*)(ar + 16 + hf * 8);   // K = 16+hf*8 .. +7
      v16bf a;
#pragma unroll
      for (int i = 0; i < 8; ++i) { a[i] = lo[i]; a[8 + i] = hi[i]; }
      afr[ti] = a;
    }
#pragma unroll
    for (int tj = 0; tj < 2; ++tj) {
      const __bf16* br = Btb[cur][wCol * 32 + tj * 16 + lm];
      v8bf lo = *(const v8bf*)(br + hf * 8);        // K = hf*8 .. +7
      v8bf hi = *(const v8bf*)(br + 16 + hf * 8);   // K = 16+hf*8 .. +7
      v16bf bvec;
#pragma unroll
      for (int i = 0; i < 8; ++i) { bvec[i] = lo[i]; bvec[8 + i] = hi[i]; }
      bfr[tj] = bvec;
    }
#pragma unroll
    for (int ti = 0; ti < 2; ++ti)
#pragma unroll
      for (int tj = 0; tj < 2; ++tj)
        acc[ti][tj] = __builtin_amdgcn_wmma_f32_16x16x32_bf16(
            false, afr[ti], false, bfr[tj], (short)0, acc[ti][tj], false, false);
    __syncthreads();
  }

#pragma unroll
  for (int ti = 0; ti < 2; ++ti)
#pragma unroll
    for (int tj = 0; tj < 2; ++tj)
#pragma unroll
      for (int r = 0; r < 8; ++r) {
        int row = m0 + wRow * 32 + ti * 16 + hf * 8 + r;
        int col = n0 + wCol * 32 + tj * 16 + lm;
        float val = acc[ti][tj][r];
        if (BIAS) val += bias[col];
        if (RELU) val = fmaxf(val, 0.0f);
        size_t ci = (size_t)row * N + col;
        if (RES) val += res[ci];
        C[ci] = (OutT)val;
      }
}

// ---------------------------------------------------------------------------
// Causal flash attention on bf16 q/k/v. One block per (b,h,128 q-rows);
// each wave owns 16 rows; uniform k-block loop (masked blocks contribute 0).
// K-tile staged via TDM (padded rows), V-tile transposed manually.
// ---------------------------------------------------------------------------
__global__ __launch_bounds__(256) void attention_kernel(
    const __bf16* __restrict__ q, const __bf16* __restrict__ k,
    const __bf16* __restrict__ v, float* __restrict__ o)
{
  __shared__ __attribute__((aligned(16))) __bf16 Kt[32][72];      // [key][hd], +8 pad
  __shared__ __attribute__((aligned(16))) __bf16 Vt[64][40];      // [hd][key], +8 pad
  __shared__ __attribute__((aligned(16))) __bf16 Pb[8][16][40];   // per-wave P

  const int tid  = threadIdx.x;
  const int lane = tid & 31;
  const int wv   = tid >> 5;
  const int lm   = lane & 15;
  const int hf   = lane >> 4;
  const int Q0 = blockIdx.x * 128;
  const int hh = blockIdx.y;
  const int bb = blockIdx.z;
  const int qrowW = Q0 + wv * 16;
  const size_t headOff = (size_t)hh * NHD;
  const float scale = 0.125f;   // 1/sqrt(64)

  // Q fragments (A layout, 16x32 per hd-chunk), direct vector loads from global
  v16bf qa[2];
  {
    const __bf16* qp = q + ((size_t)(bb * NT + qrowW + lm)) * ND + headOff;
#pragma unroll
    for (int hc = 0; hc < 2; ++hc) {
      v8bf lo = *(const v8bf*)(qp + hc * 32 + hf * 8);
      v8bf hi = *(const v8bf*)(qp + hc * 32 + 16 + hf * 8);
      v16bf a;
#pragma unroll
      for (int i = 0; i < 8; ++i) { a[i] = lo[i]; a[8 + i] = hi[i]; }
      qa[hc] = a;
    }
  }

  float mrow[8], lrow[8];
  v8f oacc[4];
#pragma unroll
  for (int r = 0; r < 8; ++r) { mrow[r] = -1e30f; lrow[r] = 0.0f; }
#pragma unroll
  for (int hg = 0; hg < 4; ++hg) oacc[hg] = zero_v8f();

  const int nkb = (Q0 + 128) >> 5;
  for (int kb = 0; kb < nkb; ++kb) {
    const int kbase = kb * 32;
#if USE_TDM
    if (wv == 0) {
      // 64 bf16/row = 32 DWORDs, then pad 4 DWORDs -> 72-elem rows
      tdm_load_2d_bf16((unsigned)(uintptr_t)&Kt[0][0],
                       k + ((size_t)(bb * NT + kbase)) * ND + headOff,
                       (unsigned)ND, (unsigned)(NB * NT), (unsigned long long)ND,
                       64u, 32u, 1u, 4u, 3u);
    }
#else
    for (int idx = tid; idx < 32 * 8; idx += 256) {
      int key = idx >> 3, c = (idx & 7) * 8;
      *(v8bf*)&Kt[key][c] =
          *(const v8bf*)(k + ((size_t)(bb * NT + kbase + key)) * ND + headOff + c);
    }
#endif
    // V transposed: Vt[hd][key]
    for (int idx = tid; idx < 32 * 64; idx += 256) {
      int key = idx >> 6, hd = idx & 63;
      Vt[hd][key] = v[((size_t)(bb * NT + kbase + key)) * ND + headOff + hd];
    }
#if USE_TDM
    if (wv == 0) __builtin_amdgcn_s_wait_tensorcnt(0);
#endif
    __syncthreads();

    // S = Q K^T : two 16x16 frags (key groups of 16), contract hd in 2 chunks
    v8f s[2];
#pragma unroll
    for (int kg = 0; kg < 2; ++kg) {
      v8f sa = zero_v8f();
#pragma unroll
      for (int hc = 0; hc < 2; ++hc) {
        const __bf16* kr = Kt[kg * 16 + lm];
        v8bf lo = *(const v8bf*)(kr + hc * 32 + hf * 8);
        v8bf hi = *(const v8bf*)(kr + hc * 32 + 16 + hf * 8);
        v16bf bk;
#pragma unroll
        for (int i = 0; i < 8; ++i) { bk[i] = lo[i]; bk[8 + i] = hi[i]; }
        sa = __builtin_amdgcn_wmma_f32_16x16x32_bf16(
            false, qa[hc], false, bk, (short)0, sa, false, false);
      }
      s[kg] = sa;
    }

    // online softmax update
    float alpha[8];
#pragma unroll
    for (int r = 0; r < 8; ++r) {
      const int rowAbs = qrowW + hf * 8 + r;
      float v0 = s[0][r] * scale;
      float v1 = s[1][r] * scale;
      if (kbase + lm > rowAbs)      v0 = -1e30f;
      if (kbase + 16 + lm > rowAbs) v1 = -1e30f;
      float rm = fmaxf(v0, v1);
#pragma unroll
      for (int off = 8; off > 0; off >>= 1) rm = fmaxf(rm, __shfl_xor(rm, off, 16));
      const float mn = fmaxf(mrow[r], rm);
      const float al = __expf(mrow[r] - mn);
      const float p0 = __expf(v0 - mn);
      const float p1 = __expf(v1 - mn);
      float ps = p0 + p1;
#pragma unroll
      for (int off = 8; off > 0; off >>= 1) ps += __shfl_xor(ps, off, 16);
      lrow[r] = lrow[r] * al + ps;
      mrow[r] = mn;
      alpha[r] = al;
      Pb[wv][hf * 8 + r][lm]      = (__bf16)p0;
      Pb[wv][hf * 8 + r][16 + lm] = (__bf16)p1;
    }
#pragma unroll
    for (int hg = 0; hg < 4; ++hg)
#pragma unroll
      for (int r = 0; r < 8; ++r) oacc[hg][r] *= alpha[r];
    __syncthreads();   // P visible for A-layout reads

    // O += P * V
    v16bf pa;
    {
      const __bf16* pr = Pb[wv][lm];
      v8bf lo = *(const v8bf*)(pr + hf * 8);
      v8bf hi = *(const v8bf*)(pr + 16 + hf * 8);
#pragma unroll
      for (int i = 0; i < 8; ++i) { pa[i] = lo[i]; pa[8 + i] = hi[i]; }
    }
#pragma unroll
    for (int hg = 0; hg < 4; ++hg) {
      const __bf16* vr = Vt[hg * 16 + lm];
      v8bf lo = *(const v8bf*)(vr + hf * 8);
      v8bf hi = *(const v8bf*)(vr + 16 + hf * 8);
      v16bf bv;
#pragma unroll
      for (int i = 0; i < 8; ++i) { bv[i] = lo[i]; bv[8 + i] = hi[i]; }
      oacc[hg] = __builtin_amdgcn_wmma_f32_16x16x32_bf16(
          false, pa, false, bv, (short)0, oacc[hg], false, false);
    }
    __syncthreads();   // tile reads done before restaging
  }

#pragma unroll
  for (int hg = 0; hg < 4; ++hg)
#pragma unroll
    for (int r = 0; r < 8; ++r) {
      const int rowAbs = qrowW + hf * 8 + r;
      const int hd = hg * 16 + lm;
      o[((size_t)(bb * NT + rowAbs)) * ND + headOff + hd] = oacc[hg][r] / lrow[r];
    }
}

// ---------------------------------------------------------------------------
// In-place softmax over vocab + round to 4 decimals (RNE), one block per row
// ---------------------------------------------------------------------------
__global__ __launch_bounds__(256) void softmax_round_kernel(float* __restrict__ logits)
{
  const int row = blockIdx.x;
  const int tid = threadIdx.x;
  float* p = logits + (size_t)row * NV;
  __shared__ float red[8];

  float m = -3.4e38f;
  for (int i = tid; i < NV; i += 256) m = fmaxf(m, p[i]);
#pragma unroll
  for (int off = 16; off > 0; off >>= 1) m = fmaxf(m, __shfl_xor(m, off, 32));
  if ((tid & 31) == 0) red[tid >> 5] = m;
  __syncthreads();
  float mm = -3.4e38f;
#pragma unroll
  for (int i = 0; i < 8; ++i) mm = fmaxf(mm, red[i]);
  __syncthreads();

  float s = 0.0f;
  for (int i = tid; i < NV; i += 256) s += expf(p[i] - mm);
#pragma unroll
  for (int off = 16; off > 0; off >>= 1) s += __shfl_xor(s, off, 32);
  if ((tid & 31) == 0) red[tid >> 5] = s;
  __syncthreads();
  float ss = 0.0f;
#pragma unroll
  for (int i = 0; i < 8; ++i) ss += red[i];
  const float inv = 1.0f / ss;

  for (int i = tid; i < NV; i += 256) {
    float pv = expf(p[i] - mm) * inv;
    p[i] = rintf(pv * 1e4f) * 1e-4f;   // round-half-even at 4 decimals
  }
}

// ---------------------------------------------------------------------------
// Launcher
// ---------------------------------------------------------------------------
extern "C" void kernel_launch(void* const* d_in, const int* in_sizes, int n_in,
                              void* d_out, int out_size, void* d_ws, size_t ws_size,
                              hipStream_t stream)
{
  (void)in_sizes; (void)n_in; (void)out_size; (void)ws_size;

  const int*   ctx   = (const int*)  d_in[0];
  const float* wte   = (const float*)d_in[1];
  const float* wpe   = (const float*)d_in[2];
  const float* Wq    = (const float*)d_in[3];
  const float* Wk    = (const float*)d_in[4];
  const float* Wv    = (const float*)d_in[5];
  const float* ln1g  = (const float*)d_in[6];
  const float* ln1b  = (const float*)d_in[7];
  const float* ln2g  = (const float*)d_in[8];
  const float* ln2b  = (const float*)d_in[9];
  const float* Wup   = (const float*)d_in[10];
  const float* bup   = (const float*)d_in[11];
  const float* Wdn   = (const float*)d_in[12];
  const float* bdn   = (const float*)d_in[13];
  const float* lnfg  = (const float*)d_in[14];
  const float* lnfb  = (const float*)d_in[15];
  const float* Wproj = (const float*)d_in[16];
  const float* bproj = (const float*)d_in[17];
  float* out = (float*)d_out;

  const size_t MT = (size_t)NB * NT;   // 2048 rows
  char* w = (char*)d_ws;
  float*  x  = (float*)w;  w += MT * ND * sizeof(float);
  float*  ob = (float*)w;  w += MT * ND * sizeof(float);
  __bf16* hB = (__bf16*)w; w += MT * ND * sizeof(__bf16);
  __bf16* qB = (__bf16*)w; w += MT * ND * sizeof(__bf16);
  __bf16* kB = (__bf16*)w; w += MT * ND * sizeof(__bf16);
  __bf16* vB = (__bf16*)w; w += MT * ND * sizeof(__bf16);
  __bf16* fB = (__bf16*)w; w += MT * (size_t)(4 * ND) * sizeof(__bf16);
  __bf16* WqT = (__bf16*)w; w += (size_t)NL * ND * ND * sizeof(__bf16);
  __bf16* WkT = (__bf16*)w; w += (size_t)NL * ND * ND * sizeof(__bf16);
  __bf16* WvT = (__bf16*)w; w += (size_t)NL * ND * ND * sizeof(__bf16);
  __bf16* WupT = (__bf16*)w; w += (size_t)NL * ND * (4 * ND) * sizeof(__bf16);
  __bf16* WdnT = (__bf16*)w; w += (size_t)NL * (4 * ND) * ND * sizeof(__bf16);
  __bf16* WprojT = (__bf16*)w;

  dim3 blk(256);
  const unsigned elemBlocks = (unsigned)((MT * ND) / 256);

  // ---- one-time (per launch) weight transpose+convert to bf16 [N][K] ----
  for (int l = 0; l < NL; ++l) {
    transpose_cvt_kernel<<<dim3(ND / 32, ND / 32), blk, 0, stream>>>(
        Wq + (size_t)l * ND * ND, WqT + (size_t)l * ND * ND, ND, ND);
    transpose_cvt_kernel<<<dim3(ND / 32, ND / 32), blk, 0, stream>>>(
        Wk + (size_t)l * ND * ND, WkT + (size_t)l * ND * ND, ND, ND);
    transpose_cvt_kernel<<<dim3(ND / 32, ND / 32), blk, 0, stream>>>(
        Wv + (size_t)l * ND * ND, WvT + (size_t)l * ND * ND, ND, ND);
    transpose_cvt_kernel<<<dim3((4 * ND) / 32, ND / 32), blk, 0, stream>>>(
        Wup + (size_t)l * ND * (4 * ND), WupT + (size_t)l * ND * (4 * ND), ND, 4 * ND);
    transpose_cvt_kernel<<<dim3(ND / 32, (4 * ND) / 32), blk, 0, stream>>>(
        Wdn + (size_t)l * (4 * ND) * ND, WdnT + (size_t)l * (4 * ND) * ND, 4 * ND, ND);
  }
  transpose_cvt_kernel<<<dim3(NV / 32, ND / 32), blk, 0, stream>>>(
      Wproj, WprojT, ND, NV);

  // ---- forward pass ----
  embed_kernel<<<dim3(elemBlocks), blk, 0, stream>>>(ctx, wte, wpe, x);

  for (int l = 0; l < NL; ++l) {
    layernorm_kernel<<<dim3((unsigned)MT), blk, 0, stream>>>(
        x, ln1g + (size_t)l * ND, ln1b + (size_t)l * ND, hB);

    dim3 gQKV(ND / 64, (unsigned)(MT / 128));
    gemm_bf16_wmma<__bf16, false, false, false><<<gQKV, blk, 0, stream>>>(
        hB, WqT + (size_t)l * ND * ND, nullptr, nullptr, qB, (int)MT, ND, ND);
    gemm_bf16_wmma<__bf16, false, false, false><<<gQKV, blk, 0, stream>>>(
        hB, WkT + (size_t)l * ND * ND, nullptr, nullptr, kB, (int)MT, ND, ND);
    gemm_bf16_wmma<__bf16, false, false, false><<<gQKV, blk, 0, stream>>>(
        hB, WvT + (size_t)l * ND * ND, nullptr, nullptr, vB, (int)MT, ND, ND);

    attention_kernel<<<dim3(NT / 128, NH, NB), blk, 0, stream>>>(qB, kB, vB, ob);
    add_kernel<<<dim3(elemBlocks), blk, 0, stream>>>(x, ob);

    layernorm_kernel<<<dim3((unsigned)MT), blk, 0, stream>>>(
        x, ln2g + (size_t)l * ND, ln2b + (size_t)l * ND, hB);

    gemm_bf16_wmma<__bf16, true, true, false>
        <<<dim3((4 * ND) / 64, (unsigned)(MT / 128)), blk, 0, stream>>>(
        hB, WupT + (size_t)l * ND * (4 * ND), bup + (size_t)l * (4 * ND), nullptr,
        fB, (int)MT, 4 * ND, ND);
    gemm_bf16_wmma<float, true, false, true>
        <<<dim3(ND / 64, (unsigned)(MT / 128)), blk, 0, stream>>>(
        fB, WdnT + (size_t)l * (4 * ND) * ND, bdn + (size_t)l * ND, x,
        x, (int)MT, ND, 4 * ND);
  }

  layernorm_kernel<<<dim3((unsigned)MT), blk, 0, stream>>>(x, lnfg, lnfb, hB);
  gemm_bf16_wmma<float, true, false, false>
      <<<dim3(NV / 64, (unsigned)(MT / 128)), blk, 0, stream>>>(
      hB, WprojT, bproj, nullptr, out, (int)MT, NV, ND);
  softmax_round_kernel<<<dim3((unsigned)MT), blk, 0, stream>>>(out);
}
